// Encoder_14594298872329
// MI455X (gfx1250) — matrix operand order repaired and checked by reading, not compiled, not run on hardware
//
#include <hip/hip_runtime.h>
#include <hip/hip_bf16.h>
#include <cstdint>
#include <cstddef>

// ---------------------------------------------------------------------------
// Bidirectional 2-layer LSTM encoder for MI455X (gfx1250, wave32, WMMA).
//   * bf16 activations/weights, f32 accumulation via v_wmma_f32_16x16x32_bf16
//   * gate-interleaved weight layout -> fused GEMM + cell/hidden update
//   * ping-pong bf16 hidden state between steps (cross-block RAW safety)
//   * one launch per timestep, fwd+bwd directions concurrent in gridDim.y
// ---------------------------------------------------------------------------

typedef __attribute__((ext_vector_type(16))) __bf16 bf16x16;
typedef __attribute__((ext_vector_type(8)))  __bf16 bf16x8;
typedef __attribute__((ext_vector_type(8)))  float  f32x8;

using u16 = unsigned short;

constexpr int B       = 128;
constexpr int MAX_LEN = 512;
constexpr int NC      = 16;
constexpr int E       = 256;
constexpr int H       = 512;
constexpr int S       = MAX_LEN + 2;   // 514
constexpr int FH      = 4 * H;         // 2048 (gate rows)
constexpr int H2      = 2 * H;         // 1024 (bi-layer output width)

// ---------------------------------------------------------------------------
// WMMA fragment loaders (ISA 7.12.2 VGPR layouts, wave32)
// A: 16x32 bf16 (M x K), source row-major [M][K], lane m = lane&15,
//    K pairs: [kh..kh+7] and [kh+16..kh+23], kh = (lane>>4)*8 -> two 16B loads
// B: 32x16 bf16 (K x N) taken from row-major weights W[N][K] (pre-transposed),
//    lane n = lane&15, K pairs: [koff..koff+7] and [koff+16..koff+23], koff=(lane>>4)*8
// ---------------------------------------------------------------------------
__device__ __forceinline__ bf16x16 load_fragA(const __bf16* base, int ld) {
  const unsigned lane = threadIdx.x & 31u;
  const unsigned r    = lane & 15u;
  const unsigned kh   = (lane >> 4) * 8u;
  const __bf16* p = base + (size_t)r * ld + kh;
  bf16x8 lo = *reinterpret_cast<const bf16x8*>(p);
  bf16x8 hi = *reinterpret_cast<const bf16x8*>(p + 16);
  return __builtin_shufflevector(lo, hi, 0,1,2,3,4,5,6,7,8,9,10,11,12,13,14,15);
}

__device__ __forceinline__ bf16x16 load_fragB(const __bf16* base, int ld) {
  const unsigned lane = threadIdx.x & 31u;
  const unsigned n    = lane & 15u;
  const unsigned koff = (lane >> 4) * 8u;
  const __bf16* p = base + (size_t)n * ld + koff;
  bf16x8 lo = *reinterpret_cast<const bf16x8*>(p);
  bf16x8 hi = *reinterpret_cast<const bf16x8*>(p + 16);
  return __builtin_shufflevector(lo, hi, 0,1,2,3,4,5,6,7,8,9,10,11,12,13,14,15);
}

__device__ __forceinline__ float sigm(float x) {
  return 1.0f / (1.0f + __expf(-x));
}

// ---------------------------------------------------------------------------
// Fused LSTM step: g = X_t @ Wih^T + h_prev @ Whh^T + bias  (gate-interleaved)
// then per-element cell/hidden update.
// grid = (FH/64 = 32, 2 dirs), block = 256 (8 waves; wave w owns batch rows
// [16w,16w+16), block owns output cols [64*bx, 64*bx+64) = 16 gate quads).
// ---------------------------------------------------------------------------
template <int KX, bool IS_L0>
__global__ __launch_bounds__(256) void lstm_step_kernel(
    const u16*  __restrict__ Xseq,    // [S][B][KX] bf16 layer input sequence
    const u16*  __restrict__ Wih,     // [2][FH][KX] bf16 (gate-interleaved)
    const u16*  __restrict__ Whh,     // [2][FH][H]  bf16 (gate-interleaved)
    const float* __restrict__ bias,   // [2][FH] f32 (bih+bhh, interleaved)
    const u16*  __restrict__ hprev,   // [2][B][H] bf16 (read parity)
    u16*        __restrict__ hnext,   // [2][B][H] bf16 (write parity)
    float*      __restrict__ h_st,    // [2][B][H] f32
    float*      __restrict__ c_st,    // [2][B][H] f32
    u16*        __restrict__ out0seq, // [S][B][2H] bf16 (layer0 only)
    int s)
{
  const int dir  = blockIdx.y;
  const int t    = (dir == 0) ? s : (S - 1 - s);
  const int n0   = blockIdx.x * 64;
  const int wave = threadIdx.x >> 5;
  const int lane = threadIdx.x & 31;
  const int m0   = wave * 16;

  const __bf16* X  = reinterpret_cast<const __bf16*>(Xseq)  + (size_t)t   * B * KX;
  const __bf16* Hp = reinterpret_cast<const __bf16*>(hprev) + (size_t)dir * B * H;
  const __bf16* Wi = reinterpret_cast<const __bf16*>(Wih)   + (size_t)dir * FH * KX;
  const __bf16* Wh = reinterpret_cast<const __bf16*>(Whh)   + (size_t)dir * FH * H;
  const float*  bs = bias + (size_t)dir * FH;

  f32x8 acc[4] = {};

  // phase 1: input contribution
  for (int k = 0; k < KX; k += 32) {
    bf16x16 a = load_fragA(X + (size_t)m0 * KX + k, KX);
#pragma unroll
    for (int j = 0; j < 4; ++j) {
      bf16x16 b = load_fragB(Wi + (size_t)(n0 + j * 16) * KX + k, KX);
      acc[j] = __builtin_amdgcn_wmma_f32_16x16x32_bf16(
          false, a, false, b, (short)0, acc[j], false, false);
    }
  }
  // phase 2: recurrent contribution
  for (int k = 0; k < H; k += 32) {
    bf16x16 a = load_fragA(Hp + (size_t)m0 * H + k, H);
#pragma unroll
    for (int j = 0; j < 4; ++j) {
      bf16x16 b = load_fragB(Wh + (size_t)(n0 + j * 16) * H + k, H);
      acc[j] = __builtin_amdgcn_wmma_f32_16x16x32_bf16(
          false, a, false, b, (short)0, acc[j], false, false);
    }
  }

  // D-matrix (ISA layout: lane col = lane&15, rows (lane>>4)*8 + r) -> LDS tile
  __shared__ float gt[8][16][64];
  const int colL  = lane & 15;
  const int rhalf = (lane >> 4) * 8;
#pragma unroll
  for (int j = 0; j < 4; ++j) {
    const int nl = j * 16 + colL;
    const float bv = bs[n0 + nl];
#pragma unroll
    for (int r = 0; r < 8; ++r)
      gt[wave][rhalf + r][nl] = acc[j][r] + bv;
  }
  __syncthreads();

  // fused pointwise: cols n0..n0+63 are 16 complete (i,f,g,o) quads
#pragma unroll
  for (int it = 0; it < 8; ++it) {
    const int idx = it * 32 + lane;
    const int row = idx >> 4;     // batch row within wave tile
    const int uu  = idx & 15;     // gate quad within block tile
    const float gi = gt[wave][row][uu * 4 + 0];
    const float gf = gt[wave][row][uu * 4 + 1];
    const float gg = gt[wave][row][uu * 4 + 2];
    const float go = gt[wave][row][uu * 4 + 3];
    const int bidx = m0 + row;
    const int u    = (n0 >> 2) + uu;   // hidden index in [0,H)
    const size_t sidx = ((size_t)dir * B + bidx) * H + u;
    const float cprev = c_st[sidx];
    const float c = sigm(gf) * cprev + sigm(gi) * tanhf(gg);
    const float h = sigm(go) * tanhf(c);
    c_st[sidx] = c;
    h_st[sidx] = h;
    reinterpret_cast<__bf16*>(hnext)[sidx] = (__bf16)h;
    if (IS_L0)
      reinterpret_cast<__bf16*>(out0seq)[((size_t)t * B + bidx) * H2 + dir * H + u] = (__bf16)h;
  }
}

// ---------------------------------------------------------------------------
// Preprocessing kernels
// ---------------------------------------------------------------------------
__global__ void scan_kernel(const int* __restrict__ lens, int* __restrict__ offs) {
  if (threadIdx.x == 0 && blockIdx.x == 0) {
    int a = 0;
    for (int b = 0; b < B; ++b) { offs[b] = a; a += lens[b]; }
  }
}

// softmax(logits) @ sym_emb + BOS/EOS/PAD assembly -> x_bf [S][B][E] bf16
__global__ __launch_bounds__(256) void embed_kernel(
    const float* __restrict__ logits, const int* __restrict__ lens,
    const int* __restrict__ offs, const float* __restrict__ sym,
    const float* __restrict__ aux, u16* __restrict__ x_bf)
{
  __shared__ float lg[NC];
  __shared__ float probs[NC];
  const int b = blockIdx.x, t = blockIdx.y, e = threadIdx.x;
  const int l = lens[b];
  const bool is_tok = (t >= 1) && (t <= l);
  if (is_tok && e < NC) {
    int row = offs[b] + t - 1;
    row = max(0, min(row, B * MAX_LEN - 1));
    lg[e] = logits[(size_t)row * NC + e];
  }
  __syncthreads();
  if (is_tok && e == 0) {
    float mx = lg[0];
    for (int c = 1; c < NC; ++c) mx = fmaxf(mx, lg[c]);
    float den = 0.f;
    for (int c = 0; c < NC; ++c) { float p = __expf(lg[c] - mx); probs[c] = p; den += p; }
    const float inv = 1.f / den;
    for (int c = 0; c < NC; ++c) probs[c] *= inv;
  }
  __syncthreads();
  float v;
  if (t == 0)          v = aux[e];
  else if (t == l + 1) v = aux[E + e];
  else if (is_tok) {
    float a = 0.f;
#pragma unroll
    for (int c = 0; c < NC; ++c) a += probs[c] * sym[c * E + e];
    v = a;
  } else               v = aux[2 * E + e];
  reinterpret_cast<__bf16*>(x_bf)[((size_t)t * B + b) * E + e] = (__bf16)v;
}

// f32 weights [2][4H][K] -> bf16, gate-interleaved rows: dst row u*4+gate
__global__ void conv_w_kernel(const float* __restrict__ src, u16* __restrict__ dst,
                              int K, size_t total)
{
  const size_t idx = (size_t)blockIdx.x * 256 + threadIdx.x;
  if (idx >= total) return;
  const int    k    = (int)(idx % K);
  const size_t r    = idx / K;
  const int    nnew = (int)(r % FH);
  const int    dir  = (int)(r / FH);
  const int    gate = nnew & 3;
  const int    u    = nnew >> 2;
  const float  v    = src[((size_t)dir * FH + gate * H + u) * K + k];
  reinterpret_cast<__bf16*>(dst)[idx] = (__bf16)v;
}

__global__ void conv_bias_kernel(const float* __restrict__ bih,
                                 const float* __restrict__ bhh,
                                 float* __restrict__ dst)
{
  const int idx = blockIdx.x * 256 + threadIdx.x;
  if (idx >= 2 * FH) return;
  const int nnew = idx % FH;
  const int dir  = idx / FH;
  const int gate = nnew & 3;
  const int u    = nnew >> 2;
  const size_t s = (size_t)dir * FH + gate * H + u;
  dst[idx] = bih[s] + bhh[s];
}

__global__ void zero_f32_kernel(float* p, size_t n) {
  const size_t i = (size_t)blockIdx.x * 256 + threadIdx.x;
  if (i < n) p[i] = 0.f;
}
__global__ void zero_u16_kernel(u16* p, size_t n) {
  const size_t i = (size_t)blockIdx.x * 256 + threadIdx.x;
  if (i < n) p[i] = 0;
}

// d_out = [h_st (4,B,H)] ++ [c_st (4,B,H)]  (state layout already matches)
__global__ void pack_kernel(const float* __restrict__ h, const float* __restrict__ c,
                            float* __restrict__ out, size_t n)
{
  const size_t i = (size_t)blockIdx.x * 256 + threadIdx.x;
  if (i < n) { out[i] = h[i]; out[n + i] = c[i]; }
}

// ---------------------------------------------------------------------------
// Host launcher
// ---------------------------------------------------------------------------
static inline size_t align_up(size_t x) { return (x + 255) & ~(size_t)255; }

extern "C" void kernel_launch(void* const* d_in, const int* in_sizes, int n_in,
                              void* d_out, int out_size, void* d_ws, size_t ws_size,
                              hipStream_t stream) {
  (void)in_sizes; (void)n_in; (void)out_size; (void)ws_size;

  const float* logits  = (const float*)d_in[0];
  const int*   lens    = (const int*)  d_in[1];
  const float* sym_emb = (const float*)d_in[2];
  const float* aux_emb = (const float*)d_in[3];
  const float* wih0    = (const float*)d_in[4];
  const float* whh0    = (const float*)d_in[5];
  const float* bih0    = (const float*)d_in[6];
  const float* bhh0    = (const float*)d_in[7];
  const float* wih1    = (const float*)d_in[8];
  const float* whh1    = (const float*)d_in[9];
  const float* bih1    = (const float*)d_in[10];
  const float* bhh1    = (const float*)d_in[11];
  float* out = (float*)d_out;

  // workspace bump allocator (~193 MB total)
  char* p = (char*)d_ws;
  auto take = [&](size_t bytes) { char* r = p; p += align_up(bytes); return r; };
  u16*   x_bf  = (u16*)  take((size_t)S * B * E  * 2);
  u16*   out0  = (u16*)  take((size_t)S * B * H2 * 2);
  u16*   wih0c = (u16*)  take((size_t)2 * FH * E  * 2);
  u16*   whh0c = (u16*)  take((size_t)2 * FH * H  * 2);
  u16*   wih1c = (u16*)  take((size_t)2 * FH * H2 * 2);
  u16*   whh1c = (u16*)  take((size_t)2 * FH * H  * 2);
  float* b0c   = (float*)take((size_t)2 * FH * 4);
  float* b1c   = (float*)take((size_t)2 * FH * 4);
  float* h_st  = (float*)take((size_t)2 * 2 * B * H * 4);   // [layer][dir][B][H]
  float* c_st  = (float*)take((size_t)2 * 2 * B * H * 4);
  u16*   h_bf  = (u16*)  take((size_t)2 * 2 * 2 * B * H * 2); // [parity][layer][dir][B][H]
  int*   offs  = (int*)  take((size_t)B * 4);

  auto nblk = [](size_t n) { return (unsigned)((n + 255) / 256); };

  // weight conversion + gate interleave (per call: deterministic, inputs untouched)
  conv_w_kernel<<<nblk((size_t)2 * FH * E),  256, 0, stream>>>(wih0, wih0c, E,  (size_t)2 * FH * E);
  conv_w_kernel<<<nblk((size_t)2 * FH * H),  256, 0, stream>>>(whh0, whh0c, H,  (size_t)2 * FH * H);
  conv_w_kernel<<<nblk((size_t)2 * FH * H2), 256, 0, stream>>>(wih1, wih1c, H2, (size_t)2 * FH * H2);
  conv_w_kernel<<<nblk((size_t)2 * FH * H),  256, 0, stream>>>(whh1, whh1c, H,  (size_t)2 * FH * H);
  conv_bias_kernel<<<nblk(2 * FH), 256, 0, stream>>>(bih0, bhh0, b0c);
  conv_bias_kernel<<<nblk(2 * FH), 256, 0, stream>>>(bih1, bhh1, b1c);

  // sequence offsets + embedding assembly
  scan_kernel<<<1, 32, 0, stream>>>(lens, offs);
  embed_kernel<<<dim3(B, S), E, 0, stream>>>(logits, lens, offs, sym_emb, aux_emb, x_bf);

  // zero initial states (c read at s=0; h_bf parity 0 read at s=0 of each layer)
  const size_t NST = (size_t)2 * 2 * B * H;
  zero_f32_kernel<<<nblk(NST), 256, 0, stream>>>(h_st, NST);
  zero_f32_kernel<<<nblk(NST), 256, 0, stream>>>(c_st, NST);
  zero_u16_kernel<<<nblk(2 * NST), 256, 0, stream>>>(h_bf, 2 * NST);

  const size_t PAR = (size_t)2 * 2 * B * H;  // parity stride (u16 elems)
  const size_t LYR = (size_t)2 * B * H;      // layer stride within parity / states

  // layer 0: fwd (t=s) and bwd (t=S-1-s) concurrently, 514 steps
  for (int s = 0; s < S; ++s) {
    const size_t pr = (size_t)(s & 1), nx = pr ^ 1;
    lstm_step_kernel<E, true><<<dim3(FH / 64, 2), 256, 0, stream>>>(
        x_bf, wih0c, whh0c, b0c,
        h_bf + pr * PAR, h_bf + nx * PAR,
        h_st, c_st, out0, s);
  }
  // layer 1 (input = full bidirectional layer-0 output)
  for (int s = 0; s < S; ++s) {
    const size_t pr = (size_t)(s & 1), nx = pr ^ 1;
    lstm_step_kernel<H2, false><<<dim3(FH / 64, 2), 256, 0, stream>>>(
        out0, wih1c, whh1c, b1c,
        h_bf + pr * PAR + LYR, h_bf + nx * PAR + LYR,
        h_st + LYR, c_st + LYR, nullptr, s);
  }

  // output: hidden (4,B,H) then cell (4,B,H) — state layout matches directly
  const size_t NOUT = (size_t)4 * B * H;
  pack_kernel<<<nblk(NOUT), 256, 0, stream>>>(h_st, c_st, out, NOUT);
}
